// CrossAttention_54666343743912
// MI455X (gfx1250) — matrix operand order repaired and checked by reading, not compiled, not run on hardware
//
#include <hip/hip_runtime.h>
#include <hip/hip_bf16.h>

#define DIM    1024
#define N_HEAD 16
#define HDIM   64

typedef __attribute__((ext_vector_type(16))) __bf16 v16bf;
typedef __attribute__((ext_vector_type(8)))  float  v8f;

// -----------------------------------------------------------------------------
// Projection GEMM:  C[M, DIM] = A[M, DIM] (f32) @ W[DIM, DIM]^T + bias
//   mode 0: outFlat f32 [M, DIM]
//   mode 1: outHead bf16 head-split   [bs, N_HEAD, len, HDIM]
//   mode 2: outHead bf16 head-split-T [bs, N_HEAD, HDIM, len]   (for V)
// 64x64 tile / 128-thread WG, K-step 64, double-buffered LDS; global loads for
// tile t+1 are register-staged and issued before the 8 WMMAs of tile t.
// -----------------------------------------------------------------------------
__global__ __launch_bounds__(128) void gemm_bf16_wmma(
    const float* __restrict__ A,
    const float* __restrict__ W,
    const float* __restrict__ bias,
    int M, float scale,
    __bf16* __restrict__ outHead, int len, int mode,
    float* __restrict__ outFlat)
{
  const int m0   = blockIdx.x * 64;
  const int n0   = blockIdx.y * 64;
  const int tid  = threadIdx.x;
  const int wave = tid >> 5;
  const int lane = tid & 31;
  const int l16  = lane & 15;
  const int hi   = lane >> 4;

  __shared__ __bf16 Asb[2][64][72];   // 64x64 bf16 tile, 144B row stride (16B aligned)
  __shared__ __bf16 Wsb[2][64][72];

  float4 ar[8], wr[8];                // register staging for one 64x64 f32 tile each

  auto load_regs = [&](int kb) {
#pragma unroll
    for (int i = 0; i < 8; ++i) {     // 8 back-to-back b128 loads per matrix
      const int c   = tid + 128 * i;  // 1024 float4 chunks
      const int row = c >> 4;
      const int c4  = (c & 15) * 4;
      ar[i] = *(const float4*)(A + (size_t)(m0 + row) * DIM + kb + c4);
      wr[i] = *(const float4*)(W + (size_t)(n0 + row) * DIM + kb + c4);
    }
  };
  auto store_lds = [&](int buf) {
#pragma unroll
    for (int i = 0; i < 8; ++i) {
      const int c   = tid + 128 * i;
      const int row = c >> 4;
      const int c4  = (c & 15) * 4;
      Asb[buf][row][c4 + 0] = (__bf16)ar[i].x;
      Asb[buf][row][c4 + 1] = (__bf16)ar[i].y;
      Asb[buf][row][c4 + 2] = (__bf16)ar[i].z;
      Asb[buf][row][c4 + 3] = (__bf16)ar[i].w;
      Wsb[buf][row][c4 + 0] = (__bf16)wr[i].x;
      Wsb[buf][row][c4 + 1] = (__bf16)wr[i].y;
      Wsb[buf][row][c4 + 2] = (__bf16)wr[i].z;
      Wsb[buf][row][c4 + 3] = (__bf16)wr[i].w;
    }
  };

  v8f zero = {};
  v8f acc[4] = {zero, zero, zero, zero};

  load_regs(0);
  store_lds(0);

  for (int kb = 0; kb < DIM; kb += 64) {
    const int  cur  = (kb >> 6) & 1;
    const bool more = (kb + 64) < DIM;
    __syncthreads();
    if (more) load_regs(kb + 64);     // global loads overlap the WMMAs below

    // A fragments 16x32 (ISA layout): lanes 0-15 K 0-7/16-23, lanes 16-31 K 8-15/24-31.
    v16bf af[2];
#pragma unroll
    for (int f = 0; f < 2; ++f)
#pragma unroll
      for (int e = 0; e < 8; ++e) {
        af[f][e]     = Asb[cur][wave * 16 + l16][f * 32 + hi * 8 + e];
        af[f][e + 8] = Asb[cur][wave * 16 + l16][f * 32 + 16 + hi * 8 + e];
      }
#pragma unroll
    for (int nt = 0; nt < 4; ++nt) {
#pragma unroll
      for (int f = 0; f < 2; ++f) {
        // B fragment 32x16: B[k][n] = W[n][k]; lane = column n, hi lanes hold K 8-15/24-31.
        v16bf bfr;
#pragma unroll
        for (int e = 0; e < 16; ++e)
          bfr[e] = Wsb[cur][nt * 16 + l16][f * 32 + (e >> 3) * 16 + hi * 8 + (e & 7)];
        acc[nt] = __builtin_amdgcn_wmma_f32_16x16x32_bf16(
            false, af[f], false, bfr, (short)0, acc[nt], false, false);
      }
    }
    if (more) store_lds(cur ^ 1);
  }

#pragma unroll
  for (int nt = 0; nt < 4; ++nt) {
#pragma unroll
    for (int r = 0; r < 8; ++r) {
      const int row = m0 + wave * 16 + r + hi * 8;   // C/D layout: vgpr r, hi lanes -> row+8
      const int col = n0 + nt * 16 + l16;
      const float v = (acc[nt][r] + bias[col]) * scale;
      if (mode == 0) {
        outFlat[(size_t)row * DIM + col] = v;
      } else {
        const int b  = row / len;
        const int qq = row - b * len;
        const int h  = col >> 6;
        const int d  = col & 63;
        if (mode == 1)
          outHead[(((size_t)b * N_HEAD + h) * len + qq) * HDIM + d] = (__bf16)v;
        else  // mode 2: transposed head layout [b, h, d, len]
          outHead[(((size_t)b * N_HEAD + h) * HDIM + d) * (size_t)len + qq] = (__bf16)v;
      }
    }
  }
}

// -----------------------------------------------------------------------------
// Flash-attention forward: one WG per (batch, head, 64-query tile); 4 waves,
// each owns a 16-query strip.  Double-buffered K/V tiles (one barrier per
// 64-key step, next tile's global loads issued before current tile's WMMAs).
// V arrives pre-transposed [b,h,HDIM,k_len], so PV B-fragments are
// row-contiguous LDS b128 reads like the K fragments.
// -----------------------------------------------------------------------------
__global__ __launch_bounds__(128) void attn_fwd_wmma(
    const __bf16* __restrict__ Qh,   // [bs,N_HEAD,q_len,HDIM], pre-scaled by 1/8
    const __bf16* __restrict__ Kh,   // [bs,N_HEAD,k_len,HDIM]
    const __bf16* __restrict__ VhT,  // [bs,N_HEAD,HDIM,k_len]
    const int* __restrict__ q_m, const int* __restrict__ k_m,
    float* __restrict__ O,           // [bs, q_len, DIM] merged heads
    int q_len, int k_len)
{
  const int q0   = blockIdx.x * 64;
  const int h    = blockIdx.y;
  const int b    = blockIdx.z;
  const int tid  = threadIdx.x;
  const int wave = tid >> 5;
  const int lane = tid & 31;
  const int l16  = lane & 15;
  const int hi   = lane >> 4;

  const __bf16* Qb = Qh  + (((size_t)b * N_HEAD + h) * q_len) * HDIM;
  const __bf16* Kb = Kh  + (((size_t)b * N_HEAD + h) * k_len) * HDIM;
  const __bf16* Vb = VhT + (((size_t)b * N_HEAD + h) * HDIM) * (size_t)k_len;

  __shared__ __bf16 Kl[2][64][72];     // keys: [key][d]
  __shared__ __bf16 Vt[2][64][72];     // values transposed: [d][key]
  __shared__ __bf16 Pl[4][16][72];     // wave-private P staging (C-layout -> A-frag)
  __shared__ int    kml[2][64];

  uint4 kreg[4], vreg[4];
  int   kmreg = 0;

  auto load_regs = [&](int kbase) {
#pragma unroll
    for (int i = 0; i < 4; ++i) {      // 512 x 16B chunks per tile
      const int c   = tid + 128 * i;
      const int row = c >> 3;
      const int c8  = (c & 7) * 8;
      kreg[i] = *(const uint4*)(Kb + (size_t)(kbase + row) * HDIM + c8);
      vreg[i] = *(const uint4*)(Vb + (size_t)row * k_len + kbase + c8);
    }
    if (tid < 64) kmreg = k_m[(size_t)b * k_len + kbase + tid];
  };
  auto store_lds = [&](int buf) {
#pragma unroll
    for (int i = 0; i < 4; ++i) {
      const int c   = tid + 128 * i;
      const int row = c >> 3;
      const int c8  = (c & 7) * 8;
      *(uint4*)&Kl[buf][row][c8] = kreg[i];
      *(uint4*)&Vt[buf][row][c8] = vreg[i];
    }
    if (tid < 64) kml[buf][tid] = kmreg;
  };

  // Q strip lives in registers as two 16x32 A-fragments (d 0-31, 32-63).
  v16bf qf[2];
  {
    const __bf16* qrow = Qb + (size_t)(q0 + wave * 16 + l16) * HDIM;
#pragma unroll
    for (int f = 0; f < 2; ++f)
#pragma unroll
      for (int e = 0; e < 8; ++e) {
        qf[f][e]     = qrow[f * 32 + hi * 8 + e];
        qf[f][e + 8] = qrow[f * 32 + 16 + hi * 8 + e];
      }
  }

  int qmk[8];
#pragma unroll
  for (int r = 0; r < 8; ++r)
    qmk[r] = q_m[(size_t)b * q_len + q0 + wave * 16 + r + hi * 8];

  float mrun[8], lrun[8];
#pragma unroll
  for (int r = 0; r < 8; ++r) { mrun[r] = -3.0e38f; lrun[r] = 0.0f; }
  v8f zero = {};
  v8f oacc[4] = {zero, zero, zero, zero};

  const int nkt = k_len / 64;
  load_regs(0);
  store_lds(0);

  for (int kt = 0; kt < nkt; ++kt) {
    const int  cur  = kt & 1;
    const bool more = (kt + 1) < nkt;
    __syncthreads();
    if (more) load_regs((kt + 1) * 64);          // overlap globals with WMMAs
    if (kt + 2 < nkt)                            // global_prefetch_b8 two tiles ahead
      __builtin_prefetch(Kb + (size_t)((kt + 2) * 64 + (tid >> 1)) * HDIM + (tid & 1) * 32, 0, 3);

    // ---- S = Q K^T for this wave's 16x64 strip ----
    v8f s[4] = {zero, zero, zero, zero};
#pragma unroll
    for (int nt = 0; nt < 4; ++nt) {
#pragma unroll
      for (int f = 0; f < 2; ++f) {
        v16bf kfr;
#pragma unroll
        for (int e = 0; e < 16; ++e)
          kfr[e] = Kl[cur][nt * 16 + l16][f * 32 + (e >> 3) * 16 + hi * 8 + (e & 7)];
        s[nt] = __builtin_amdgcn_wmma_f32_16x16x32_bf16(
            false, qf[f], false, kfr, (short)0, s[nt], false, false);
      }
    }

    // ---- mask + online softmax (a score row = one VGPR across 16 lanes) ----
#pragma unroll
    for (int r = 0; r < 8; ++r) {
      float mx = -3.0e38f;
#pragma unroll
      for (int nt = 0; nt < 4; ++nt) {
        const int keep = qmk[r] & kml[cur][nt * 16 + l16];
        const float v = keep ? s[nt][r] : -3.0e38f;
        s[nt][r] = v;
        mx = fmaxf(mx, v);
      }
#pragma unroll
      for (int off = 1; off < 16; off <<= 1)
        mx = fmaxf(mx, __shfl_xor(mx, off, 32));
      const float nm   = fmaxf(mrun[r], mx);
      const float corr = __expf(mrun[r] - nm);
      float rs = 0.0f;
#pragma unroll
      for (int nt = 0; nt < 4; ++nt) {
        const float p = __expf(s[nt][r] - nm);
        s[nt][r] = p;
        rs += p;
      }
#pragma unroll
      for (int off = 1; off < 16; off <<= 1)
        rs += __shfl_xor(rs, off, 32);
      lrun[r] = lrun[r] * corr + rs;
      mrun[r] = nm;
#pragma unroll
      for (int nt = 0; nt < 4; ++nt) oacc[nt][r] *= corr;
    }

    // ---- re-stage P via wave-private LDS (C/D layout -> A-fragment layout) ----
#pragma unroll
    for (int nt = 0; nt < 4; ++nt)
#pragma unroll
      for (int r = 0; r < 8; ++r)
        Pl[wave][r + hi * 8][nt * 16 + l16] = (__bf16)s[nt][r];

    v16bf pf[2];
#pragma unroll
    for (int f = 0; f < 2; ++f)
#pragma unroll
      for (int e = 0; e < 8; ++e) {
        pf[f][e]     = Pl[wave][l16][f * 32 + hi * 8 + e];
        pf[f][e + 8] = Pl[wave][l16][f * 32 + 16 + hi * 8 + e];
      }

    // ---- O += P V : B[k][d] = Vt[d][k] -> row-contiguous LDS reads ----
#pragma unroll
    for (int dt = 0; dt < 4; ++dt) {
#pragma unroll
      for (int f = 0; f < 2; ++f) {
        v16bf vfr;
#pragma unroll
        for (int e = 0; e < 16; ++e)
          vfr[e] = Vt[cur][dt * 16 + l16][f * 32 + (e >> 3) * 16 + hi * 8 + (e & 7)];
        oacc[dt] = __builtin_amdgcn_wmma_f32_16x16x32_bf16(
            false, pf[f], false, vfr, (short)0, oacc[dt], false, false);
      }
    }

    if (more) store_lds(cur ^ 1);
  }

  // ---- normalize + merge heads: O[b, q, h*64 + d] ----
#pragma unroll
  for (int dt = 0; dt < 4; ++dt)
#pragma unroll
    for (int r = 0; r < 8; ++r) {
      const int row = q0 + wave * 16 + r + hi * 8;
      const int col = h * HDIM + dt * 16 + l16;
      O[((size_t)b * q_len + row) * DIM + col] = oacc[dt][r] / lrun[r];
    }
}

// -----------------------------------------------------------------------------
// Launch: 3 projection GEMMs -> attention -> output GEMM.
// Workspace: Qh bf16 4MB | Kh bf16 16MB | VhT bf16 16MB | O f32 8MB  (44MB)
// -----------------------------------------------------------------------------
extern "C" void kernel_launch(void* const* d_in, const int* in_sizes, int n_in,
                              void* d_out, int out_size, void* d_ws, size_t ws_size,
                              hipStream_t stream) {
  (void)in_sizes; (void)n_in; (void)out_size; (void)ws_size;
  const float* q   = (const float*)d_in[0];
  const float* k   = (const float*)d_in[1];
  const int*   q_m = (const int*)d_in[2];
  const int*   k_m = (const int*)d_in[3];
  const float* Wq  = (const float*)d_in[4];
  const float* bq  = (const float*)d_in[5];
  const float* Wk  = (const float*)d_in[6];
  const float* bk  = (const float*)d_in[7];
  const float* Wv  = (const float*)d_in[8];
  const float* bv  = (const float*)d_in[9];
  const float* Wo  = (const float*)d_in[10];
  const float* bo  = (const float*)d_in[11];
  float* out = (float*)d_out;

  const int bs = 2, q_len = 1024, k_len = 4096;
  const int Mq = bs * q_len;   // 2048
  const int Mk = bs * k_len;   // 8192

  char* ws = (char*)d_ws;
  __bf16* Qws = (__bf16*)(ws);
  __bf16* Kws = (__bf16*)(ws + (4ull  << 20));
  __bf16* Vws = (__bf16*)(ws + (20ull << 20));
  float*  Ows = (float*) (ws + (36ull << 20));

  dim3 blk(128);
  // Q projection, fused bias + 1/sqrt(HDIM) scale, bf16 head-split output.
  gemm_bf16_wmma<<<dim3(Mq / 64, DIM / 64), blk, 0, stream>>>(
      q, Wq, bq, Mq, 0.125f, Qws, q_len, 1, nullptr);
  gemm_bf16_wmma<<<dim3(Mk / 64, DIM / 64), blk, 0, stream>>>(
      k, Wk, bk, Mk, 1.0f, Kws, k_len, 1, nullptr);
  // V projection written transposed per head so attention reads it row-major.
  gemm_bf16_wmma<<<dim3(Mk / 64, DIM / 64), blk, 0, stream>>>(
      k, Wv, bv, Mk, 1.0f, Vws, k_len, 2, nullptr);

  attn_fwd_wmma<<<dim3(q_len / 64, N_HEAD, bs), blk, 0, stream>>>(
      Qws, Kws, Vws, q_m, k_m, Ows, q_len, k_len);

  // Output projection: f32 flat into d_out.
  gemm_bf16_wmma<<<dim3(Mq / 64, DIM / 64), blk, 0, stream>>>(
      Ows, Wo, bo, Mq, 1.0f, nullptr, 1, 0, out);
}
